// CSABlock_72292889526809
// MI455X (gfx1250) — matrix operand order repaired and checked by reading, not compiled, not run on hardware
//
#include <hip/hip_runtime.h>
#include <hip/hip_bf16.h>

// ---------------------------------------------------------------------------
// CSA spatial-attention block, fused for MI455X (gfx1250, wave32, WMMA).
// Shapes (hardcoded from reference): B=2, C=16, D=80, H=96, W=96, K=7, PAD=3.
// ---------------------------------------------------------------------------

typedef _Float16 v16h __attribute__((ext_vector_type(16)));
typedef _Float16 v4h  __attribute__((ext_vector_type(4)));
typedef float    v8f  __attribute__((ext_vector_type(8)));

#define CSA_B   2
#define CSA_C   16
#define CSA_D   80
#define CSA_H   96
#define CSA_W   96
#define CSA_HW  (CSA_H * CSA_W)           // 9216
#define CSA_DHW (CSA_D * CSA_H * CSA_W)   // 737280

// ---------------------------------------------------------------------------
// Kernel 0: expand conv taps into WMMA A-fragment lane layout (one shot).
// A[m,j] = taps[c,dz,dy][j-m-1] for (j-m-1) in [0,6], else 0.
// A lane layout used here (contiguous k, same as B): element e <-> K index j = e + blo,
// with m = lane&15, blo = (lane>>4)*16.
// afrag layout: [g(2)][combo = (dz*2+c)*7+dy (98)][lane(32)][16 halves]
// ---------------------------------------------------------------------------
__global__ void csa_expand_weights(const float* __restrict__ w_f2m,
                                   const float* __restrict__ w_m2f,
                                   _Float16* __restrict__ afrag) {
    const int t = blockIdx.x * blockDim.x + threadIdx.x;
    if (t >= 2 * 98 * 32) return;
    const int lane  = t & 31;
    const int combo = (t >> 5) % 98;
    const int g     = t / (98 * 32);

    const int dy = combo % 7;
    const int zc = combo / 7;          // dz*2 + c
    const int c  = zc & 1;
    const int dz = zc >> 1;

    const float* wsrc = g ? w_m2f : w_f2m;
    const float* taps = wsrc + ((c * 7 + dz) * 7 + dy) * 7;

    const int m   = lane & 15;
    const int blo = (lane >> 4) * 16;

    v16h a;
#pragma unroll
    for (int e = 0; e < 16; ++e) {
        const int j = e + blo;
        const unsigned dx = (unsigned)(j - m - 1);
        a[e] = (dx <= 6u) ? (_Float16)taps[dx] : (_Float16)0;
    }
    *(v16h*)&afrag[((size_t)(g * 98 + combo) * 32 + lane) * 16] = a;
}

// ---------------------------------------------------------------------------
// Kernel 1: channel max + mean pooling, fp32 -> f16 pooled planes.
// grid.y = 0 -> fix, 1 -> move. Each thread handles 4 consecutive voxels.
// ---------------------------------------------------------------------------
__global__ void csa_pool_kernel(const float* __restrict__ fix,
                                const float* __restrict__ move,
                                _Float16* __restrict__ pooledF,
                                _Float16* __restrict__ pooledM) {
    const int tid = blockIdx.x * blockDim.x + threadIdx.x; // < B*DHW/4
    const int i4  = tid * 4;
    const int b   = i4 / CSA_DHW;
    const int r   = i4 - b * CSA_DHW;

    const float* x = (blockIdx.y ? move : fix) + (size_t)(b * CSA_C) * CSA_DHW + r;
    _Float16* pooled = (blockIdx.y ? pooledM : pooledF);

    float mx0 = -3.4e38f, mx1 = -3.4e38f, mx2 = -3.4e38f, mx3 = -3.4e38f;
    float s0 = 0.f, s1 = 0.f, s2 = 0.f, s3 = 0.f;
#pragma unroll
    for (int c = 0; c < CSA_C; ++c) {
        const float4 v = *(const float4*)(x + (size_t)c * CSA_DHW);
        mx0 = fmaxf(mx0, v.x); mx1 = fmaxf(mx1, v.y);
        mx2 = fmaxf(mx2, v.z); mx3 = fmaxf(mx3, v.w);
        s0 += v.x; s1 += v.y; s2 += v.z; s3 += v.w;
    }
    const float inv = 1.0f / (float)CSA_C;
    v4h mxh = { (_Float16)mx0, (_Float16)mx1, (_Float16)mx2, (_Float16)mx3 };
    v4h avh = { (_Float16)(s0 * inv), (_Float16)(s1 * inv),
                (_Float16)(s2 * inv), (_Float16)(s3 * inv) };
    *(v4h*)&pooled[(size_t)(b * 2 + 0) * CSA_DHW + r] = mxh;  // max channel
    *(v4h*)&pooled[(size_t)(b * 2 + 1) * CSA_DHW + r] = avh;  // avg channel
}

// ---------------------------------------------------------------------------
// Kernel 2: 7x7x7 conv(2ch -> 1) + sigmoid, banded-Toeplitz f16 WMMA.
//
// One 64-thread WG (2 waves) produces a 16(w) x 32(h) gate tile at fixed
// (gate, b, d0). Per wave: D[m,n] (m = w offset, n = h offset) accumulated
// over 98 WMMAs (dz, c, dy):
//   A[m,j] = taps[c,dz,dy][j-m-1]                  (precomputed fragments)
//   B[j,n] = pooled[c, d0+dz-3, h0+wave*16+n+dy-3, w0-4+j]  (LDS halo block)
// Valid band j in [m+1, m+7] subset [1,22]; all other K slots multiply exact
// zeros in A, so the staged 32-wide window needs no special K padding.
// ---------------------------------------------------------------------------
__global__ void csa_conv_wmma_kernel(const _Float16* __restrict__ pooledF,
                                     const _Float16* __restrict__ pooledM,
                                     const _Float16* __restrict__ afrag,
                                     float* __restrict__ gateF,
                                     float* __restrict__ gateM) {
    // plane: [dz(7)][c(2)][row(38)][col(32)] f16 = 34,048 B
    __shared__ _Float16 plane[7 * 2 * 38 * 32];

    int id = blockIdx.x;
    const int wt = id % 6;  id /= 6;     // w tile (16 wide)
    const int ht = id % 3;  id /= 3;     // h tile (32 tall)
    const int d0 = id % CSA_D; id /= CSA_D;
    const int b  = id % CSA_B; id /= CSA_B;
    const int g  = id;                   // 0: fix-gate (w_f2m), 1: move-gate (w_m2f)

    const int h0 = ht * 32;
    const int w0 = wt * 16;

    const _Float16* pooled = g ? pooledM : pooledF;
    const _Float16* afragG = afrag + (size_t)g * 98 * 32 * 16;
    float*          gate   = g ? gateM : gateF;

    const int tid  = threadIdx.x;        // 0..63
    const int lane = tid & 31;
    const int wave = tid >> 5;           // 0..1 -> h sub-tile

    // ---- stage pooled halo block: 532 rows x 8 aligned quads (b64) ---------
    // Window base w0-4 makes every 4-half quad 8B-aligned and fully in or
    // fully out of [0, 96).
    const int q = tid & 7;               // quad within row
    const int wq = w0 - 4 + q * 4;
    const bool win = ((unsigned)wq < (unsigned)CSA_W);
    for (int rowIdx = (tid >> 3); rowIdx < 532; rowIdx += 8) {
        const int dz  = rowIdx / 76;
        const int rem = rowIdx - dz * 76;
        const int c   = rem / 38;
        const int rr  = rem - c * 38;
        const int d   = d0 + dz - 3;
        const int h   = h0 + rr - 3;
        v4h v = { (_Float16)0, (_Float16)0, (_Float16)0, (_Float16)0 };
        if (win && (unsigned)d < (unsigned)CSA_D && (unsigned)h < (unsigned)CSA_H) {
            v = *(const v4h*)&pooled[(size_t)((b * 2 + c) * CSA_D + d) * CSA_HW +
                                     h * CSA_W + wq];
        }
        *(v4h*)&plane[((dz * 2 + c) * 38 + rr) * 32 + q * 4] = v;
    }
    __syncthreads();

    // ---- WMMA accumulation --------------------------------------------------
    const int ncol  = lane & 15;         // N = h offset within wave tile
    const int khalf = lane >> 4;         // B-fragment K half (K = khalf*16 + e)
    const int hh    = h0 + wave * 16;

    v8f acc = { 0.f, 0.f, 0.f, 0.f, 0.f, 0.f, 0.f, 0.f };

    for (int dz = 0; dz < 7; ++dz) {
#pragma unroll
        for (int c = 0; c < 2; ++c) {
#pragma unroll
            for (int dy = 0; dy < 7; ++dy) {
                const int combo = (dz * 2 + c) * 7 + dy;
                // A: precomputed Toeplitz fragment (32B coalesced, L2-hot)
                const v16h a = *(const v16h*)&afragG[((size_t)combo * 32 + lane) * 16];
                // B: contiguous 16 halves (32B) per lane from the staged plane
                const v16h bfrag = *(const v16h*)&plane[
                    ((dz * 2 + c) * 38 + (wave * 16 + ncol + dy)) * 32 + khalf * 16];

                acc = __builtin_amdgcn_wmma_f32_16x16x32_f16(
                    /*neg_a=*/false, a, /*neg_b=*/false, bfrag,
                    /*c_mod=*/(short)0, acc, /*reuse_a=*/false, /*reuse_b=*/false);
            }
        }
    }

    // ---- sigmoid + store. C/D layout: lanes 0-15 -> M=r, lanes 16-31 -> M=8+r;
    // N = lane&15. M = w offset, N = h offset.
    const int hrow = hh + ncol;
    float* gout = gate + (size_t)(b * CSA_D + d0) * CSA_HW + hrow * CSA_W + w0;
#pragma unroll
    for (int r = 0; r < 8; ++r) {
        const float v = acc[r];
        gout[r + khalf * 8] = 1.0f / (1.0f + __expf(-v));
    }
}

// ---------------------------------------------------------------------------
// Kernel 3: fused gating, float4-vectorized.
//   fix_out = move * gateF + fix ; move_out = fix * gateM + move
// ---------------------------------------------------------------------------
__global__ void csa_combine_kernel(const float* __restrict__ fix,
                                   const float* __restrict__ move,
                                   const float* __restrict__ gateF,
                                   const float* __restrict__ gateM,
                                   float* __restrict__ out) {
    const int idx = blockIdx.x * blockDim.x + threadIdx.x; // < B*C*DHW/4
    const int i4  = idx * 4;
    const int bc  = i4 / CSA_DHW;        // b*16 + c
    const int r   = i4 - bc * CSA_DHW;
    const int b   = bc >> 4;
    const size_t gidx = (size_t)b * CSA_DHW + r;

    const float4 f  = *(const float4*)(fix  + (size_t)i4);
    const float4 mv = *(const float4*)(move + (size_t)i4);
    const float4 gF = *(const float4*)(gateF + gidx);
    const float4 gM = *(const float4*)(gateM + gidx);

    float4 of, om;
    of.x = fmaf(mv.x, gF.x, f.x);  om.x = fmaf(f.x, gM.x, mv.x);
    of.y = fmaf(mv.y, gF.y, f.y);  om.y = fmaf(f.y, gM.y, mv.y);
    of.z = fmaf(mv.z, gF.z, f.z);  om.z = fmaf(f.z, gM.z, mv.z);
    of.w = fmaf(mv.w, gF.w, f.w);  om.w = fmaf(f.w, gM.w, mv.w);

    const size_t N4 = (size_t)CSA_B * CSA_C * CSA_DHW / 4; // 5,898,240
    ((float4*)out)[idx]      = of;
    ((float4*)out)[idx + N4] = om;
}

// ---------------------------------------------------------------------------
extern "C" void kernel_launch(void* const* d_in, const int* in_sizes, int n_in,
                              void* d_out, int out_size, void* d_ws, size_t ws_size,
                              hipStream_t stream) {
    (void)in_sizes; (void)n_in; (void)out_size; (void)ws_size;

    const float* fix   = (const float*)d_in[0];
    const float* move  = (const float*)d_in[1];
    const float* w_f2m = (const float*)d_in[2];
    const float* w_m2f = (const float*)d_in[3];
    float* out = (float*)d_out;

    // Workspace layout (23,793,664 bytes total):
    //   pooledF : f16 [B][2][D][H][W]        (5,898,240 B)
    //   pooledM : f16 [B][2][D][H][W]        (5,898,240 B)
    //   gateF   : f32 [B][D][H][W]           (5,898,240 B)
    //   gateM   : f32 [B][D][H][W]           (5,898,240 B)
    //   afrag   : f16 [2][98][32][16]        (  200,704 B)
    char* ws = (char*)d_ws;
    _Float16* pooledF = (_Float16*)ws;
    _Float16* pooledM = pooledF + (size_t)CSA_B * 2 * CSA_DHW;
    float* gateF = (float*)(ws + 2ull * CSA_B * 2 * CSA_DHW * sizeof(_Float16));
    float* gateM = gateF + (size_t)CSA_B * CSA_DHW;
    _Float16* afrag = (_Float16*)(gateM + (size_t)CSA_B * CSA_DHW);

    // 0) expand conv taps into WMMA A-fragment layout (tiny, one shot)
    csa_expand_weights<<<(2 * 98 * 32 + 127) / 128, 128, 0, stream>>>(
        w_f2m, w_m2f, afrag);

    // 1) pooling: B*DHW/4 threads per tensor, grid.y selects fix/move
    {
        dim3 grid((CSA_B * CSA_DHW / 4 + 255) / 256, 2);
        csa_pool_kernel<<<grid, 256, 0, stream>>>(fix, move, pooledF, pooledM);
    }
    // 2) conv + sigmoid via WMMA: 2 gates * B * D * 3(ht) * 6(wt) WGs of 64
    {
        dim3 grid(2 * CSA_B * CSA_D * 3 * 6);
        csa_conv_wmma_kernel<<<grid, 64, 0, stream>>>(pooledF, pooledM, afrag,
                                                      gateF, gateM);
    }
    // 3) fused gating
    {
        dim3 grid((CSA_B * CSA_C * CSA_DHW / 4 + 255) / 256);
        csa_combine_kernel<<<grid, 256, 0, stream>>>(fix, move, gateF, gateM, out);
    }
}
